// FullAttention_38783554683221
// MI455X (gfx1250) — matrix operand order repaired and checked by reading, not compiled, not run on hardware
//
#include <hip/hip_runtime.h>

// Flash-attention for B=4,H=16,S=2048,D=64 fp32 (no 1/sqrt(d) scale, non-causal).
// Matmuls on v_wmma_f32_16x16x32_f16 (f16 inputs, f32 accumulate).
// Row-max via fused v_max_num_f32_dpp butterfly; row-sum via a 9th WMMA (P x ones).
// Global K/V loads register-double-buffered to overlap HBM latency with compute.

typedef __attribute__((ext_vector_type(16))) _Float16 v16h;
typedef __attribute__((ext_vector_type(8)))  _Float16 v8h;
typedef __attribute__((ext_vector_type(8)))  float    v8f;

#define SLEN  2048
#define DHEAD 64
#define JTILE 32     // key tile width per iteration
#define NWAVE 8      // waves per block; each wave owns 16 query rows
#define KSTR  72     // LDS K row stride (halves): conflict-free + 16B aligned
#define VSTR  40     // LDS V^T / P row stride (halves)

union H16 { v16h v; v8h h[2]; };

// Plain max without fmaxf's canonicalize ops (inputs are finite here).
__device__ __forceinline__ float vmaxf(float a, float b) {
  float d;
  asm("v_max_num_f32 %0, %1, %2" : "=v"(d) : "v"(a), "v"(b));
  return d;
}

// Max-reduce across the 16 lanes of each DPP row (wave32: two independent
// halves, matching the WMMA fragment lane groups). Each butterfly step is a
// single fused v_max_num_f32_dpp (src0 permuted, src1 direct).
__device__ __forceinline__ float rowmax16(float t) {
  float a, b, c, d;
  asm("v_max_num_f32_dpp %0, %1, %1 quad_perm:[1,0,3,2] row_mask:0xf bank_mask:0xf bound_ctrl:1"
      : "=&v"(a) : "v"(t));
  asm("v_max_num_f32_dpp %0, %1, %1 quad_perm:[2,3,0,1] row_mask:0xf bank_mask:0xf bound_ctrl:1"
      : "=&v"(b) : "v"(a));
  asm("v_max_num_f32_dpp %0, %1, %1 row_half_mirror row_mask:0xf bank_mask:0xf bound_ctrl:1"
      : "=&v"(c) : "v"(b));
  asm("v_max_num_f32_dpp %0, %1, %1 row_mirror row_mask:0xf bank_mask:0xf bound_ctrl:1"
      : "=&v"(d) : "v"(c));
  return d;
}

__global__ __launch_bounds__(256) void fa_fwd_f32in(const float* __restrict__ Q,
                                                    const float* __restrict__ K,
                                                    const float* __restrict__ V,
                                                    float* __restrict__ O) {
  __shared__ alignas(16) _Float16 ldsK [JTILE][KSTR];        // K tile  [j][d]
  __shared__ alignas(16) _Float16 ldsVt[DHEAD][VSTR];        // V tile transposed [d][j]
  __shared__ alignas(16) _Float16 ldsP [NWAVE][16][VSTR];    // per-wave P 16x32

  const int tid  = threadIdx.x;
  const int wave = tid >> 5;
  const int lane = tid & 31;
  const int n    = lane & 15;   // fragment column
  const int hs   = lane >> 4;   // half-select (lane group)

  const int bh   = blockIdx.x >> 4;        // 0..63  (b*H+h)
  const int qblk = blockIdx.x & 15;        // 16 q-blocks of 128 rows
  const int q0   = qblk * (16 * NWAVE) + wave * 16;
  const size_t base = (size_t)bh * SLEN * DHEAD;

  const float LOG2E = 1.44269504088896f;

  // ---- Q * log2e -> two f16 A-fragments (K-dim chunks d0..31, d32..63) ----
  // A layout: lane holds row m = lane&15; element e -> d = e + 16*hs
  const float* qp = Q + base + (size_t)(q0 + n) * DHEAD;
  H16 a0u, a1u;
  #pragma unroll
  for (int e = 0; e < 8; ++e) {
    a0u.h[0][e] = (_Float16)(qp[      16*hs + e] * LOG2E);
    a0u.h[1][e] = (_Float16)(qp[8 +  16*hs + e] * LOG2E);
    a1u.h[0][e] = (_Float16)(qp[32 +  16*hs + e] * LOG2E);
    a1u.h[1][e] = (_Float16)(qp[40 +  16*hs + e] * LOG2E);
  }
  const v16h qa0 = a0u.v, qa1 = a1u.v;

  // All-ones B fragment: P x ones accumulates the softmax denominator l.
  H16 onesu;
  #pragma unroll
  for (int e = 0; e < 8; ++e) { onesu.h[0][e] = (_Float16)1.0f; onesu.h[1][e] = (_Float16)1.0f; }
  const v16h onesB = onesu.v;

  // Straight-line staging assignment: thread owns key row sj, d-cols sd..sd+7.
  const int sj = tid >> 3;          // 0..31
  const int sd = (tid & 7) << 3;    // 0,8,...,56
  const float* kPtr = K + base + (size_t)sj * DHEAD + sd;
  const float* vPtr = V + base + (size_t)sj * DHEAD + sd;

  // softmax state: rows r + 8*hs (replicated across each 16-lane half)
  float mrow[8];
  v8f accO[4], lacc;
  #pragma unroll
  for (int r = 0; r < 8; ++r) mrow[r] = -3.0e38f;
  #pragma unroll
  for (int c = 0; c < 4; ++c) accO[c] = (v8f){};
  lacc = (v8f){};

  // ---- prologue: load tile 0 into registers (double buffer) ----
  float4 ka = *(const float4*)(kPtr);
  float4 kb = *(const float4*)(kPtr + 4);
  float4 va = *(const float4*)(vPtr);
  float4 vb = *(const float4*)(vPtr + 4);

  for (int j0 = 0; j0 < SLEN; j0 += JTILE) {
    __syncthreads();  // previous iteration's LDS consumers are done

    // ---- convert + store the registered tile (K packed b128, V transposed) ----
    {
      v8h kh;
      kh[0] = (_Float16)ka.x; kh[1] = (_Float16)ka.y;
      kh[2] = (_Float16)ka.z; kh[3] = (_Float16)ka.w;
      kh[4] = (_Float16)kb.x; kh[5] = (_Float16)kb.y;
      kh[6] = (_Float16)kb.z; kh[7] = (_Float16)kb.w;
      *(v8h*)&ldsK[sj][sd] = kh;
      ldsVt[sd + 0][sj] = (_Float16)va.x; ldsVt[sd + 1][sj] = (_Float16)va.y;
      ldsVt[sd + 2][sj] = (_Float16)va.z; ldsVt[sd + 3][sj] = (_Float16)va.w;
      ldsVt[sd + 4][sj] = (_Float16)vb.x; ldsVt[sd + 5][sj] = (_Float16)vb.y;
      ldsVt[sd + 6][sj] = (_Float16)vb.z; ldsVt[sd + 7][sj] = (_Float16)vb.w;
    }
    __syncthreads();

    // ---- issue next tile's global loads now; they overlap all compute below.
    // Branchless tail: re-load tile 0 on the last iteration (harmless).
    {
      const size_t off = (j0 + JTILE < SLEN) ? (size_t)(j0 + JTILE) * DHEAD : 0;
      ka = *(const float4*)(kPtr + off);
      kb = *(const float4*)(kPtr + off + 4);
      va = *(const float4*)(vPtr + off);
      vb = *(const float4*)(vPtr + off + 4);
    }

    // ---- S' = (Q*log2e) K^T : two 16x16 column-subtiles, f32 accumulate ----
    // B layout: lane holds column (= key row) n; element e -> k(d) = e + 16*hs
    v8f sfr[2];
    #pragma unroll
    for (int s = 0; s < 2; ++s) {
      H16 b0u, b1u;
      b0u.h[0] = *(const v8h*)&ldsK[s*16 + n][16*hs];
      b0u.h[1] = *(const v8h*)&ldsK[s*16 + n][16*hs + 8];
      b1u.h[0] = *(const v8h*)&ldsK[s*16 + n][32 + 16*hs];
      b1u.h[1] = *(const v8h*)&ldsK[s*16 + n][32 + 16*hs + 8];
      v8f z = (v8f){};
      v8f acc = __builtin_amdgcn_wmma_f32_16x16x32_f16(false, qa0, false, b0u.v,
                                                       (short)0, z, false, false);
      acc     = __builtin_amdgcn_wmma_f32_16x16x32_f16(false, qa1, false, b1u.v,
                                                       (short)0, acc, false, false);
      sfr[s] = acc;
    }

    // ---- online softmax in exp2 domain (scores already scaled by log2e) ----
    #pragma unroll
    for (int r = 0; r < 8; ++r) {
      const float t     = rowmax16(vmaxf(sfr[0][r], sfr[1][r]));
      const float mnew  = vmaxf(mrow[r], t);
      const float scale = __builtin_amdgcn_exp2f(mrow[r] - mnew);
      const float p0    = __builtin_amdgcn_exp2f(sfr[0][r] - mnew);
      const float p1    = __builtin_amdgcn_exp2f(sfr[1][r] - mnew);
      mrow[r] = mnew;
      lacc[r] *= scale;
      #pragma unroll
      for (int c = 0; c < 4; ++c) accO[c][r] *= scale;
      // D-layout -> LDS: row = r + 8*hs, cols n and 16+n
      ldsP[wave][r + 8*hs][n]      = (_Float16)p0;
      ldsP[wave][r + 8*hs][16 + n] = (_Float16)p1;
    }

    // ldsP is wave-private: same-wave LDS ops retire in order (DScnt in-order),
    // and may-alias addressing keeps compiler order. Only block reordering here;
    // no workgroup barrier needed (waves are resynced by the staging barriers).
    asm volatile("" ::: "memory");

    // ---- P as A-fragment: lane row m = n; element e -> j = e+16*hs ----
    H16 pu;
    pu.h[0] = *(const v8h*)&ldsP[wave][n][16*hs];
    pu.h[1] = *(const v8h*)&ldsP[wave][n][16*hs + 8];
    const v16h pa = pu.v;

    // ---- l += P * ones (row sums, replicated across columns) ----
    lacc = __builtin_amdgcn_wmma_f32_16x16x32_f16(false, pa, false, onesB,
                                                  (short)0, lacc, false, false);

    // ---- O += P * V : B from transposed V, lane column d = c*16+n, k(j)=e+16*hs ----
    #pragma unroll
    for (int c = 0; c < 4; ++c) {
      H16 vb2;
      vb2.h[0] = *(const v8h*)&ldsVt[c*16 + n][16*hs];
      vb2.h[1] = *(const v8h*)&ldsVt[c*16 + n][16*hs + 8];
      accO[c] = __builtin_amdgcn_wmma_f32_16x16x32_f16(false, pa, false, vb2.v,
                                                       (short)0, accO[c], false, false);
    }
  }

  // ---- epilogue: divide by l, store fp32 ----
  float inv[8];
  #pragma unroll
  for (int r = 0; r < 8; ++r) inv[r] = 1.0f / lacc[r];
  #pragma unroll
  for (int c = 0; c < 4; ++c) {
    #pragma unroll
    for (int r = 0; r < 8; ++r) {
      const int row = r + 8*hs;
      O[base + (size_t)(q0 + row) * DHEAD + c*16 + n] = accO[c][r] * inv[r];
    }
  }
}

extern "C" void kernel_launch(void* const* d_in, const int* in_sizes, int n_in,
                              void* d_out, int out_size, void* d_ws, size_t ws_size,
                              hipStream_t stream) {
  (void)in_sizes; (void)n_in; (void)out_size; (void)d_ws; (void)ws_size;
  const float* q = (const float*)d_in[0];
  const float* k = (const float*)d_in[1];
  const float* v = (const float*)d_in[2];
  float* out = (float*)d_out;
  // grid = B*H * (S/128) = 64 * 16 = 1024 blocks of 256 threads (8 waves)
  fa_fwd_f32in<<<dim3(1024), dim3(256), 0, stream>>>(q, k, v, out);
}
